// EncoderCell_49624052138102
// MI455X (gfx1250) — matrix operand order repaired and checked by reading, not compiled, not run on hardware
//
#include <hip/hip_runtime.h>
#include <hip/hip_bf16.h>

// ---------------------------------------------------------------------------
// GRU cell on MI455X (gfx1250): two-phase bf16 WMMA, double-buffered LDS.
//   Phase 1: Ar = x@Wr^T + h@Ur^T, Az = x@Wz^T + h@Uz^T   (fused K = 3072)
//            -> rh = sigmoid(Ar+br) * hprev  (bf16, ws)
//            -> z  = sigmoid(Az+bz)          (bf16, ws)
//   Phase 2: S = x@Wh^T + rh@Uh^T
//            -> out = z*tanh(S+bh) + (1-z)*hprev
// Block tile 128x128, 8 waves (2 M x 4 N), wave tile 64x32 per gate:
// 16 WMMA per K-step per wave vs 16 ds_load_b128 (1:1).
// ---------------------------------------------------------------------------

typedef __attribute__((ext_vector_type(16))) __bf16 v16bf;
typedef __attribute__((ext_vector_type(8)))  float  v8f;

#define BATCH 4096
#define IN    1024
#define HID   2048

#define BM 128           // block tile rows (batch)
#define BN 128           // block tile cols (hidden)
#define BK 32            // K step (WMMA K for bf16)
#define LDH 40           // LDS row stride in halves (32 + 8 pad -> 80B = 20 banks)

__device__ __forceinline__ float fast_sigmoid(float v) {
    return 1.0f / (1.0f + __expf(-v));
}
__device__ __forceinline__ float fast_tanh(float v) {
    return 1.0f - 2.0f / (__expf(2.0f * v) + 1.0f);
}

__device__ __forceinline__ v8f wmma_bf16(v16bf a, v16bf b, v8f c) {
    return __builtin_amdgcn_wmma_f32_16x16x32_bf16(
        false, a, false, b, (short)0, c, false, false);
}

// A-fragment: 16x32 bf16, rows (M) striped across lanes.
// lanes 0-15 -> M=lane, K in {0..7, 16..23}; lanes 16-31 -> M=lane-16, K {8..15, 24..31}.
__device__ __forceinline__ v16bf load_frag_a(const __bf16* lds, int row0, int lane) {
    const int m  = row0 + (lane & 15);
    const int kb = (lane >> 4) << 3;          // 0 or 8
    const __bf16* p = lds + m * LDH + kb;
    v16bf f;
    ((uint4*)&f)[0] = *(const uint4*)(p);
    ((uint4*)&f)[1] = *(const uint4*)(p + 16);
    return f;
}

// B-fragment: 32x16 where B[k][n] = W[n][k]; W tile row-major (N x K) in LDS.
// lanes 0-15 -> N=lane, K in {0..7, 16..23}; lanes 16-31 -> N=lane-16, K {8..15, 24..31}.
__device__ __forceinline__ v16bf load_frag_b(const __bf16* lds, int row0, int lane) {
    const int n  = row0 + (lane & 15);
    const int kb = (lane >> 4) << 3;          // 0 or 8
    const __bf16* p = lds + n * LDH + kb;
    v16bf f;
    ((uint4*)&f)[0] = *(const uint4*)(p);
    ((uint4*)&f)[1] = *(const uint4*)(p + 16);
    return f;
}

__device__ __forceinline__ void cvt_store4(__bf16* d, float4 v) {
    union { __bf16 h[4]; uint2 u; } pk;
    pk.h[0] = (__bf16)v.x; pk.h[1] = (__bf16)v.y;
    pk.h[2] = (__bf16)v.z; pk.h[3] = (__bf16)v.w;
    *(uint2*)d = pk.u;
}

// ===========================================================================
// Phase 1
// ===========================================================================
struct P1Regs { float4 a[4], b0[4], b1[4]; };

__device__ __forceinline__ void p1_load(P1Regs& s,
    const float* __restrict__ A, int rowBase,
    const float* __restrict__ B0, const float* __restrict__ B1, int colBase,
    int ldg, int kk, int tid) {
#pragma unroll
    for (int i = 0; i < 4; ++i) {
        const int idx = tid + (i << 8);       // 0..1023 (128x32 f32 / float4)
        const int r = idx >> 3, c = (idx & 7) << 2;
        s.a[i]  = *(const float4*)(A  + (size_t)(rowBase + r) * ldg + kk + c);
        s.b0[i] = *(const float4*)(B0 + (size_t)(colBase + r) * ldg + kk + c);
        s.b1[i] = *(const float4*)(B1 + (size_t)(colBase + r) * ldg + kk + c);
    }
}

__device__ __forceinline__ void p1_store(__bf16* sA, __bf16* sB0, __bf16* sB1,
                                         const P1Regs& s, int tid) {
#pragma unroll
    for (int i = 0; i < 4; ++i) {
        const int idx = tid + (i << 8);
        const int r = idx >> 3, c = (idx & 7) << 2;
        cvt_store4(sA  + r * LDH + c, s.a[i]);
        cvt_store4(sB0 + r * LDH + c, s.b0[i]);
        cvt_store4(sB1 + r * LDH + c, s.b1[i]);
    }
}

__device__ __forceinline__ void p1_compute(const __bf16* sA, const __bf16* sB0,
                                           const __bf16* sB1,
                                           v8f accR[4][2], v8f accZ[4][2],
                                           int wm, int wn, int lane) {
    v16bf a[4], b0[2], b1[2];
#pragma unroll
    for (int i = 0; i < 4; ++i) a[i] = load_frag_a(sA, wm * 64 + i * 16, lane);
#pragma unroll
    for (int j = 0; j < 2; ++j) {
        b0[j] = load_frag_b(sB0, wn * 32 + j * 16, lane);
        b1[j] = load_frag_b(sB1, wn * 32 + j * 16, lane);
    }
#pragma unroll
    for (int i = 0; i < 4; ++i)
#pragma unroll
        for (int j = 0; j < 2; ++j) {
            accR[i][j] = wmma_bf16(a[i], b0[j], accR[i][j]);
            accZ[i][j] = wmma_bf16(a[i], b1[j], accZ[i][j]);
        }
}

__global__ __launch_bounds__(256) void gru_phase1(
    const float* __restrict__ x,  const float* __restrict__ hprev,
    const float* __restrict__ Wz, const float* __restrict__ Wr,
    const float* __restrict__ Uz, const float* __restrict__ Ur,
    const float* __restrict__ pbz, const float* __restrict__ pbr,
    __bf16* __restrict__ rh_ws, __bf16* __restrict__ z_ws)
{
    __shared__ __bf16 sm[2][3][BM * LDH];   // double-buffered A, Br, Bz

    const int tid  = threadIdx.x;
    const int lane = tid & 31;
    const int wave = tid >> 5;
    const int wm   = wave & 1;                 // 2 waves in M (64 rows each)
    const int wn   = wave >> 1;                // 4 waves in N (32 cols each)
    const int rowBase = blockIdx.y * BM;
    const int colBase = blockIdx.x * BN;

    v8f accR[4][2] = {};
    v8f accZ[4][2] = {};
    P1Regs regs;

    // prologue: stage K-step 0 (x segment)
    p1_load(regs, x, rowBase, Wr, Wz, colBase, IN, 0, tid);
    p1_store(sm[0][0], sm[0][1], sm[0][2], regs, tid);
    int buf = 0;

    const int KTOT = IN + HID;               // fused K = 3072
    for (int k0 = BK; k0 < KTOT; k0 += BK) {
        const bool second = (k0 >= IN);
        const float* A  = second ? hprev : x;
        const float* B0 = second ? Ur : Wr;
        const float* B1 = second ? Uz : Wz;
        const int ldg   = second ? HID : IN;
        const int kk    = second ? (k0 - IN) : k0;
        p1_load(regs, A, rowBase, B0, B1, colBase, ldg, kk, tid); // prefetch k0
        __syncthreads();                                          // sm[buf] ready
        p1_compute(sm[buf][0], sm[buf][1], sm[buf][2], accR, accZ, wm, wn, lane);
        p1_store(sm[buf ^ 1][0], sm[buf ^ 1][1], sm[buf ^ 1][2], regs, tid);
        buf ^= 1;
    }
    __syncthreads();
    p1_compute(sm[buf][0], sm[buf][1], sm[buf][2], accR, accZ, wm, wn, lane);

    // epilogue: gates + elementwise products
    const int halfSel = (lane >> 4) << 3;
    const int nl      = lane & 15;
#pragma unroll
    for (int j = 0; j < 2; ++j) {
        const int col = colBase + wn * 32 + j * 16 + nl;
        const float biasR = pbr[col];
        const float biasZ = pbz[col];
#pragma unroll
        for (int i = 0; i < 4; ++i) {
#pragma unroll
            for (int e = 0; e < 8; ++e) {
                const int row = rowBase + wm * 64 + i * 16 + halfSel + e;
                const size_t o = (size_t)row * HID + col;
                const float rv = fast_sigmoid(accR[i][j][e] + biasR);
                const float zv = fast_sigmoid(accZ[i][j][e] + biasZ);
                const float hp = hprev[o];
                rh_ws[o] = (__bf16)(rv * hp);
                z_ws[o]  = (__bf16)zv;
            }
        }
    }
}

// ===========================================================================
// Phase 2
// ===========================================================================
struct P2RegsF { float4 a[4], b[4]; };       // A from f32 (x)
struct P2RegsH { uint2  a[4]; float4 b[4]; };// A from bf16 (rh_ws)

__device__ __forceinline__ void p2_load_f(P2RegsF& s,
    const float* __restrict__ A, int rowBase,
    const float* __restrict__ B, int colBase, int ldg, int kk, int tid) {
#pragma unroll
    for (int i = 0; i < 4; ++i) {
        const int idx = tid + (i << 8);
        const int r = idx >> 3, c = (idx & 7) << 2;
        s.a[i] = *(const float4*)(A + (size_t)(rowBase + r) * ldg + kk + c);
        s.b[i] = *(const float4*)(B + (size_t)(colBase + r) * ldg + kk + c);
    }
}

__device__ __forceinline__ void p2_load_h(P2RegsH& s,
    const __bf16* __restrict__ A, int rowBase,
    const float* __restrict__ B, int colBase, int kk, int tid) {
#pragma unroll
    for (int i = 0; i < 4; ++i) {
        const int idx = tid + (i << 8);
        const int r = idx >> 3, c = (idx & 7) << 2;
        s.a[i] = *(const uint2*)(A + (size_t)(rowBase + r) * HID + kk + c);
        s.b[i] = *(const float4*)(B + (size_t)(colBase + r) * HID + kk + c);
    }
}

__device__ __forceinline__ void p2_store_f(__bf16* sA, __bf16* sB,
                                           const P2RegsF& s, int tid) {
#pragma unroll
    for (int i = 0; i < 4; ++i) {
        const int idx = tid + (i << 8);
        const int r = idx >> 3, c = (idx & 7) << 2;
        cvt_store4(sA + r * LDH + c, s.a[i]);
        cvt_store4(sB + r * LDH + c, s.b[i]);
    }
}

__device__ __forceinline__ void p2_store_h(__bf16* sA, __bf16* sB,
                                           const P2RegsH& s, int tid) {
#pragma unroll
    for (int i = 0; i < 4; ++i) {
        const int idx = tid + (i << 8);
        const int r = idx >> 3, c = (idx & 7) << 2;
        *(uint2*)(sA + r * LDH + c) = s.a[i];
        cvt_store4(sB + r * LDH + c, s.b[i]);
    }
}

__device__ __forceinline__ void p2_compute(const __bf16* sA, const __bf16* sB,
                                           v8f acc[4][2], int wm, int wn, int lane) {
    v16bf a[4], b[2];
#pragma unroll
    for (int i = 0; i < 4; ++i) a[i] = load_frag_a(sA, wm * 64 + i * 16, lane);
#pragma unroll
    for (int j = 0; j < 2; ++j) b[j] = load_frag_b(sB, wn * 32 + j * 16, lane);
#pragma unroll
    for (int i = 0; i < 4; ++i)
#pragma unroll
        for (int j = 0; j < 2; ++j)
            acc[i][j] = wmma_bf16(a[i], b[j], acc[i][j]);
}

__global__ __launch_bounds__(256) void gru_phase2(
    const float* __restrict__ x,  const float* __restrict__ hprev,
    const float* __restrict__ Wh, const float* __restrict__ Uh,
    const float* __restrict__ pbh,
    const __bf16* __restrict__ rh_ws, const __bf16* __restrict__ z_ws,
    float* __restrict__ out)
{
    __shared__ __bf16 sm[2][2][BM * LDH];   // double-buffered A, B

    const int tid  = threadIdx.x;
    const int lane = tid & 31;
    const int wave = tid >> 5;
    const int wm   = wave & 1;
    const int wn   = wave >> 1;
    const int rowBase = blockIdx.y * BM;
    const int colBase = blockIdx.x * BN;

    v8f acc[4][2] = {};
    P2RegsF rf;
    P2RegsH rh;

    // ---- segment 1: x @ Wh^T, double buffered
    p2_load_f(rf, x, rowBase, Wh, colBase, IN, 0, tid);
    p2_store_f(sm[0][0], sm[0][1], rf, tid);
    int buf = 0;
    for (int k0 = BK; k0 < IN; k0 += BK) {
        p2_load_f(rf, x, rowBase, Wh, colBase, IN, k0, tid);
        __syncthreads();
        p2_compute(sm[buf][0], sm[buf][1], acc, wm, wn, lane);
        p2_store_f(sm[buf ^ 1][0], sm[buf ^ 1][1], rf, tid);
        buf ^= 1;
    }
    // boundary handoff into segment 2 (A dtype changes to bf16)
    p2_load_h(rh, rh_ws, rowBase, Uh, colBase, 0, tid);
    __syncthreads();
    p2_compute(sm[buf][0], sm[buf][1], acc, wm, wn, lane);
    p2_store_h(sm[buf ^ 1][0], sm[buf ^ 1][1], rh, tid);
    buf ^= 1;

    // ---- segment 2: (r*hprev) @ Uh^T, double buffered
    for (int k0 = BK; k0 < HID; k0 += BK) {
        p2_load_h(rh, rh_ws, rowBase, Uh, colBase, k0, tid);
        __syncthreads();
        p2_compute(sm[buf][0], sm[buf][1], acc, wm, wn, lane);
        p2_store_h(sm[buf ^ 1][0], sm[buf ^ 1][1], rh, tid);
        buf ^= 1;
    }
    __syncthreads();
    p2_compute(sm[buf][0], sm[buf][1], acc, wm, wn, lane);

    // ---- epilogue: out = z*tanh(S+bh) + (1-z)*hprev
    const int halfSel = (lane >> 4) << 3;
    const int nl      = lane & 15;
#pragma unroll
    for (int j = 0; j < 2; ++j) {
        const int col = colBase + wn * 32 + j * 16 + nl;
        const float biasH = pbh[col];
#pragma unroll
        for (int i = 0; i < 4; ++i) {
#pragma unroll
            for (int e = 0; e < 8; ++e) {
                const int row = rowBase + wm * 64 + i * 16 + halfSel + e;
                const size_t o = (size_t)row * HID + col;
                const float t  = fast_tanh(acc[i][j][e] + biasH);
                const float zv = (float)z_ws[o];
                const float hp = hprev[o];
                out[o] = hp + zv * (t - hp);
            }
        }
    }
}

// ---------------------------------------------------------------------------
extern "C" void kernel_launch(void* const* d_in, const int* in_sizes, int n_in,
                              void* d_out, int out_size, void* d_ws, size_t ws_size,
                              hipStream_t stream) {
    // setup_inputs order: x, hprev, Wh, Wz, Wr, Uh, Uz, Ur, bh, bz, br
    const float* x     = (const float*)d_in[0];
    const float* hprev = (const float*)d_in[1];
    const float* Wh    = (const float*)d_in[2];
    const float* Wz    = (const float*)d_in[3];
    const float* Wr    = (const float*)d_in[4];
    const float* Uh    = (const float*)d_in[5];
    const float* Uz    = (const float*)d_in[6];
    const float* Ur    = (const float*)d_in[7];
    const float* bh    = (const float*)d_in[8];
    const float* bz    = (const float*)d_in[9];
    const float* br    = (const float*)d_in[10];
    float* out = (float*)d_out;

    // Workspace: [rh bf16 : 16MB][z bf16 : 16MB] = 32MB total.
    __bf16* rh_ws = (__bf16*)d_ws;
    __bf16* z_ws  = rh_ws + (size_t)BATCH * HID;

    dim3 grid(HID / BN, BATCH / BM);   // 16 x 32
    dim3 block(256);

    gru_phase1<<<grid, block, 0, stream>>>(x, hprev, Wz, Wr, Uz, Ur, bz, br,
                                           rh_ws, z_ws);
    gru_phase2<<<grid, block, 0, stream>>>(x, hprev, Wh, Uh, bh,
                                           rh_ws, z_ws, out);
}
